// CorticalSheet_33260226740445
// MI455X (gfx1250) — compile-verified, matrix operand order checked by reading, not run on hardware
//
#include <hip/hip_runtime.h>
#include <stdint.h>

// ---------------- CDNA5 WMMA types ----------------
typedef __attribute__((ext_vector_type(16))) __bf16      bf16x16;
typedef __attribute__((ext_vector_type(8)))  float       v8f;
typedef __attribute__((ext_vector_type(8)))  unsigned    v8u;
typedef __attribute__((ext_vector_type(4)))  int         i32x4;

// ---------------- constants ----------------
#define B_   256
#define T_   16
#define D_   2048
#define C_   16
#define K_   1024
#define MROWS (B_*T_)          // 4096

#define OUT_RES   0
#define OUT_OP    524288
#define OUT_PRIOR 534784
#define OUT_GATE  535552
#define OUT_SCAL  539648

// GEMM tiling
#define BM 128
#define BN 128
#define BK 32
#define APITCH 20        // dwords per LDS row: 16 data + 4 pad (80B, 16B-aligned)

// ---------------- helpers ----------------
__device__ __forceinline__ unsigned short f2bf(float x) {
    unsigned u = __builtin_bit_cast(unsigned, x);
    unsigned r = u + 0x7FFFu + ((u >> 16) & 1u);   // RNE
    return (unsigned short)(r >> 16);
}
__device__ __forceinline__ float bf2f(unsigned short u) {
    return __builtin_bit_cast(float, (unsigned)u << 16);
}
// gelu(x) = 0.5x(1+tanh(u)) = x * sigmoid(2u), u = sqrt(2/pi)(x + 0.044715 x^3)
__device__ __forceinline__ float gelu_fast(float x) {
    const float k0 = 1.5957691216057308f;   // 2*sqrt(2/pi)
    const float k1 = 0.044715f;
    float u = k0 * (x + k1 * x * x * x);
    return x / (1.f + __expf(-u));
}
__device__ __forceinline__ float sigmoid_fast(float x) {
    return 1.f / (1.f + __expf(-x));
}
__device__ __forceinline__ float block_sum256(float v, float* sh) {
    int tid = threadIdx.x;
    sh[tid] = v; __syncthreads();
    #pragma unroll
    for (int s = 128; s > 0; s >>= 1) { if (tid < s) sh[tid] += sh[tid + s]; __syncthreads(); }
    float r = sh[0]; __syncthreads();
    return r;
}
__device__ __forceinline__ float block_max256(float v, float* sh) {
    int tid = threadIdx.x;
    sh[tid] = v; __syncthreads();
    #pragma unroll
    for (int s = 128; s > 0; s >>= 1) { if (tid < s) sh[tid] = fmaxf(sh[tid], sh[tid + s]); __syncthreads(); }
    float r = sh[0]; __syncthreads();
    return r;
}

// ---------------- CDNA5 async global->LDS copy (ASYNCcnt path) ----------------
#if defined(__has_builtin)
#  if __has_builtin(__builtin_amdgcn_global_load_async_to_lds_b128)
#    define HAVE_ASYNC_BUILTIN 1
#  endif
#  if __has_builtin(__builtin_amdgcn_s_wait_asynccnt)
#    define HAVE_WAIT_ASYNC_BUILTIN 1
#  endif
#endif

typedef __attribute__((address_space(1))) i32x4 gas_i32x4;
typedef __attribute__((address_space(3))) i32x4 las_i32x4;

__device__ __forceinline__ void async_copy16(const void* g, void* l) {
#ifdef HAVE_ASYNC_BUILTIN
    __builtin_amdgcn_global_load_async_to_lds_b128(
        (gas_i32x4*)g, (las_i32x4*)l, 0, 0);
#else
    unsigned lo = (unsigned)(uintptr_t)l;
    asm volatile("global_load_async_to_lds_b128 %0, %1, off"
                 :: "v"(lo), "v"(g) : "memory");
#endif
}
__device__ __forceinline__ void wait_async0() {
#ifdef HAVE_WAIT_ASYNC_BUILTIN
    __builtin_amdgcn_s_wait_asynccnt(0);
#else
    asm volatile("s_wait_asynccnt 0x0" ::: "memory");
#endif
}

// ---------------- pre-pass: f32 -> bf16 convert ----------------
__global__ __launch_bounds__(256) void cvt_bf16_kernel(
    const float* __restrict__ x, unsigned short* __restrict__ y)
{
    int i = (blockIdx.x * 256 + threadIdx.x) * 4;
    const float4 f = *(const float4*)(x + i);
    y[i + 0] = f2bf(f.x); y[i + 1] = f2bf(f.y);
    y[i + 2] = f2bf(f.z); y[i + 3] = f2bf(f.w);
}

// ---------------- pre-pass: W[Kd][N] f32 -> Wt[N][Kd] bf16 (batched) ----------------
__global__ __launch_bounds__(256) void transpose_bf16_kernel(
    const float* __restrict__ W, unsigned short* __restrict__ Wt, int Kd, int N)
{
    __shared__ float tile[32][33];
    const float*    Wb  = W  + (size_t)blockIdx.z * Kd * N;
    unsigned short* Wtb = Wt + (size_t)blockIdx.z * Kd * N;
    int k0 = blockIdx.x * 32, n0 = blockIdx.y * 32;
    int tx = threadIdx.x & 31, ty = threadIdx.x >> 5;   // 32 x 8
    #pragma unroll
    for (int i = 0; i < 4; ++i)
        tile[ty + i * 8][tx] = Wb[(size_t)(k0 + ty + i * 8) * N + n0 + tx];
    __syncthreads();
    #pragma unroll
    for (int i = 0; i < 4; ++i)
        Wtb[(size_t)(n0 + ty + i * 8) * Kd + k0 + tx] = f2bf(tile[tx][ty + i * 8]);
}

// ---------------- WMMA GEMM: out = epilogue(A[MxKd] * Wt[NxKd]^T + bias) ----------------
// A, Wt bf16. Double-buffered async staging: copies for tile k+1 overlap WMMA on tile k.
// EP==0: bf16 out = gelu(v) ; EP==1: f32 out = v - tokens
template<int EP>
__global__ __launch_bounds__(256) void gemm_bf16_ep(
    const unsigned short* __restrict__ A,  const unsigned short* __restrict__ Wt,
    const float* __restrict__ bias, const float* __restrict__ tok,
    unsigned short* __restrict__ outb, float* __restrict__ outf,
    int M, int N, int Kd)
{
    __shared__ unsigned lA[2][BM * APITCH];
    __shared__ unsigned lB[2][BN * APITCH];

    const int tid  = threadIdx.x;
    const int lane = tid & 31;
    const int w    = tid >> 5;
    const int wm   = w >> 1;          // 0..3 : 32-row slab
    const int wn   = w & 1;           // 0..1 : 64-col slab
    const int hf   = lane >> 4;
    const int l16  = lane & 15;

    const int mBase = blockIdx.y * BM;
    const int nBase = blockIdx.x * BN;

    auto stage = [&](int buf, int kb) {
        #pragma unroll
        for (int it = 0; it < 2; ++it) {
            int slot = tid + it * 256;            // 0..511
            int row  = slot >> 2;                 // 0..127
            int q    = slot & 3;                  // 16B chunk within 64B row
            async_copy16(A  + (size_t)(mBase + row) * Kd + kb + q * 8,
                         &lA[buf][row * APITCH + q * 4]);
            async_copy16(Wt + (size_t)(nBase + row) * Kd + kb + q * 8,
                         &lB[buf][row * APITCH + q * 4]);
        }
    };

    v8f acc[2][4];
    #pragma unroll
    for (int mi = 0; mi < 2; ++mi)
        #pragma unroll
        for (int ni = 0; ni < 4; ++ni)
            acc[mi][ni] = v8f{0.f,0.f,0.f,0.f,0.f,0.f,0.f,0.f};

    stage(0, 0);
    int cur = 0;
    for (int kb = 0; kb < Kd; kb += BK) {
        wait_async0();        // current buffer's async copies landed in LDS
        __syncthreads();      // all waves' copies complete; prior reads of other buf done
        if (kb + BK < Kd) stage(cur ^ 1, kb + BK);   // overlap next copies with WMMA

        // ---- fragments per CDNA5 16-bit 16x32 layout ----
        v8u au[2], bu[4];
        #pragma unroll
        for (int mi = 0; mi < 2; ++mi) {
            int r = wm * 32 + mi * 16 + l16;
            #pragma unroll
            for (int p = 0; p < 8; ++p) {
                int kp = (p < 4 ? p : p + 4) + hf * 4;
                au[mi][p] = lA[cur][r * APITCH + kp];
            }
        }
        #pragma unroll
        for (int ni = 0; ni < 4; ++ni) {
            int cn = wn * 64 + ni * 16 + l16;
            #pragma unroll
            for (int p = 0; p < 8; ++p) {
                int kp = (p < 4 ? p : p + 4) + hf * 4;
                bu[ni][p] = lB[cur][cn * APITCH + kp];
            }
        }
        #pragma unroll
        for (int mi = 0; mi < 2; ++mi) {
            bf16x16 av = __builtin_bit_cast(bf16x16, au[mi]);
            #pragma unroll
            for (int ni = 0; ni < 4; ++ni) {
                bf16x16 bv = __builtin_bit_cast(bf16x16, bu[ni]);
                acc[mi][ni] = __builtin_amdgcn_wmma_f32_16x16x32_bf16(
                    false, av, false, bv, (short)0, acc[mi][ni], false, false);
            }
        }
        cur ^= 1;
    }

    // ---- epilogue: C/D layout lane->N, vgpr r -> M = r + 8*hf ----
    #pragma unroll
    for (int mi = 0; mi < 2; ++mi) {
        #pragma unroll
        for (int ni = 0; ni < 4; ++ni) {
            int n = nBase + wn * 64 + ni * 16 + l16;
            float bz = bias[n];
            #pragma unroll
            for (int r = 0; r < 8; ++r) {
                int m = mBase + wm * 32 + mi * 16 + hf * 8 + r;
                float v = acc[mi][ni][r] + bz;
                size_t idx = (size_t)m * N + n;
                if (EP == 0) outb[idx] = f2bf(gelu_fast(v));
                else         outf[idx] = v - tok[idx];
            }
        }
    }
}

// ---------------- gate: LN(mean_t tokens) @ gw + gb, top-4 masked softmax ----------------
__global__ __launch_bounds__(256) void gate_kernel(
    const float* __restrict__ tokens, const float* __restrict__ gs, const float* __restrict__ gbv,
    const float* __restrict__ gw, const float* __restrict__ gbias, float* __restrict__ gate_out)
{
    __shared__ float sh[256];
    __shared__ float red[256 * 16];
    int b = blockIdx.x, tid = threadIdx.x;
    float x[8]; float s1 = 0.f, s2 = 0.f;
    #pragma unroll
    for (int i = 0; i < 8; ++i) {
        int d = tid + i * 256;
        float a = 0.f;
        for (int t = 0; t < T_; ++t) a += tokens[((size_t)b * T_ + t) * D_ + d];
        x[i] = a * (1.f / T_);
        s1 += x[i]; s2 += x[i] * x[i];
    }
    float t1 = block_sum256(s1, sh), t2 = block_sum256(s2, sh);
    float mean = t1 * (1.f / D_);
    float var  = t2 * (1.f / D_) - mean * mean;
    float rs   = rsqrtf(var + 1e-5f);
    float part[C_];
    #pragma unroll
    for (int c = 0; c < C_; ++c) part[c] = 0.f;
    #pragma unroll
    for (int i = 0; i < 8; ++i) {
        int d = tid + i * 256;
        float ln = (x[i] - mean) * rs * gs[d] + gbv[d];
        #pragma unroll
        for (int c = 0; c < C_; ++c) part[c] += ln * gw[(size_t)d * C_ + c];
    }
    #pragma unroll
    for (int c = 0; c < C_; ++c) red[tid * C_ + c] = part[c];
    __syncthreads();
    for (int s = 128; s > 0; s >>= 1) {
        if (tid < s) {
            #pragma unroll
            for (int c = 0; c < C_; ++c) red[tid * C_ + c] += red[(tid + s) * C_ + c];
        }
        __syncthreads();
    }
    if (tid == 0) {
        float lgv[C_]; bool used[C_]; float mask[C_];
        for (int c = 0; c < C_; ++c) { lgv[c] = red[c] + gbias[c]; used[c] = false; mask[c] = 0.f; }
        for (int j = 0; j < 4; ++j) {               // TOPK = 4, ties -> lowest index
            int best = -1; float bv = -1e30f;
            for (int c = 0; c < C_; ++c) if (!used[c] && lgv[c] > bv) { bv = lgv[c]; best = c; }
            used[best] = true; mask[best] = 1.f;
        }
        float mx = -1e30f;
        for (int c = 0; c < C_; ++c) mx = fmaxf(mx, lgv[c]);
        float se = 0.f, e[C_];
        for (int c = 0; c < C_; ++c) { e[c] = __expf(lgv[c] - mx); se += e[c]; }
        float gsum = 0.f, gp[C_];
        for (int c = 0; c < C_; ++c) { gp[c] = (e[c] / se) * mask[c]; gsum += gp[c]; }
        float inv = 1.f / fmaxf(gsum, 1e-6f);
        for (int c = 0; c < C_; ++c) gate_out[b * C_ + c] = gp[c] * inv;
    }
}

// ---------------- per-column reductions ----------------
__global__ __launch_bounds__(256) void reduce_err_kernel(
    const float* __restrict__ err, float* __restrict__ res_c, float* __restrict__ sse)
{
    __shared__ float sh[256];
    int idx = blockIdx.x * 256 + threadIdx.x;       // over B*D
    int b = idx >> 11, d = idx & (D_ - 1);
    float s = 0.f, sq = 0.f;
    #pragma unroll
    for (int t = 0; t < T_; ++t) {
        float e = err[((size_t)(b * T_ + t)) * D_ + d];
        s += e; sq += e * e;
    }
    res_c[idx] = s * (1.f / T_);
    float tot = block_sum256(sq, sh);
    if (threadIdx.x == 0) atomicAdd(sse, tot);
}

__global__ __launch_bounds__(256) void zred_kernel(
    const unsigned short* __restrict__ h3, float* __restrict__ z_c)
{
    int idx = blockIdx.x * 256 + threadIdx.x;       // over B*K
    int b = idx >> 10, k = idx & (K_ - 1);
    float s = 0.f;
    #pragma unroll
    for (int t = 0; t < T_; ++t) s += bf2f(h3[((size_t)(b * T_ + t)) * K_ + k]);
    z_c[idx] = s * (1.f / T_);
}

__global__ __launch_bounds__(256) void entropy_kernel(
    const unsigned short* __restrict__ h3, float* __restrict__ ent_sum)
{
    __shared__ float sh[256];
    int m = blockIdx.x, tid = threadIdx.x;
    float v[4]; float mx = -1e30f;
    #pragma unroll
    for (int i = 0; i < 4; ++i) {
        v[i] = bf2f(h3[(size_t)m * K_ + tid + i * 256]);
        mx = fmaxf(mx, v[i]);
    }
    mx = block_max256(mx, sh);
    float se = 0.f, sx = 0.f;
    #pragma unroll
    for (int i = 0; i < 4; ++i) {
        float e = __expf(v[i] - mx);
        se += e; sx += (v[i] - mx) * e;
    }
    se = block_sum256(se, sh);
    sx = block_sum256(sx, sh);
    if (tid == 0) atomicAdd(ent_sum, __logf(se) - sx / se);
}

// ---------------- gated combine ----------------
__global__ __launch_bounds__(256) void combine_residual_kernel(
    const float* __restrict__ gate, const float* __restrict__ res, float* __restrict__ out)
{
    int idx = blockIdx.x * 256 + threadIdx.x;       // over B*D
    int b = idx >> 11;
    float a = 0.f;
    #pragma unroll
    for (int c = 0; c < C_; ++c) a += gate[b * C_ + c] * res[(size_t)c * (B_ * D_) + idx];
    out[idx] = a;
}
__global__ __launch_bounds__(256) void combine_latent_kernel(
    const float* __restrict__ gate, const float* __restrict__ z, float* __restrict__ lat)
{
    int idx = blockIdx.x * 256 + threadIdx.x;       // over B*K
    int b = idx >> 10;
    float a = 0.f;
    #pragma unroll
    for (int c = 0; c < C_; ++c) a += gate[b * C_ + c] * z[(size_t)c * (B_ * K_) + idx];
    lat[idx] = a;
}

// ---------------- heads ----------------
__global__ __launch_bounds__(256) void op_head_kernel(
    const float* __restrict__ lat, const float* __restrict__ os, const float* __restrict__ ob,
    const float* __restrict__ w1, const float* __restrict__ b1,
    const float* __restrict__ w2, const float* __restrict__ b2, float* __restrict__ outp)
{
    __shared__ float sh[256];
    __shared__ float ln[K_];
    __shared__ float hid[128];
    __shared__ float lgt[48];
    __shared__ float mxse[2];
    int b = blockIdx.x, tid = threadIdx.x;
    float v[4]; float s1 = 0.f, s2 = 0.f;
    #pragma unroll
    for (int i = 0; i < 4; ++i) {
        int k = tid + i * 256;
        v[i] = lat[(size_t)b * K_ + k];
        s1 += v[i]; s2 += v[i] * v[i];
    }
    float t1 = block_sum256(s1, sh), t2 = block_sum256(s2, sh);
    float mean = t1 * (1.f / K_), var = t2 * (1.f / K_) - mean * mean;
    float rs = rsqrtf(var + 1e-5f);
    #pragma unroll
    for (int i = 0; i < 4; ++i) {
        int k = tid + i * 256;
        ln[k] = (v[i] - mean) * rs * os[k] + ob[k];
    }
    __syncthreads();
    if (tid < 128) {
        float a = b1[tid];
        for (int k = 0; k < K_; ++k) a += ln[k] * w1[(size_t)k * 128 + tid];
        hid[tid] = gelu_fast(a);
    }
    __syncthreads();
    if (tid < 41) {
        float a = b2[tid];
        #pragma unroll
        for (int j = 0; j < 128; ++j) a += hid[j] * w2[j * 41 + tid];
        lgt[tid] = a;
    }
    __syncthreads();
    if (tid == 0) {
        float mx = -1e30f;
        for (int t = 0; t < 41; ++t) mx = fmaxf(mx, lgt[t]);
        float se = 0.f;
        for (int t = 0; t < 41; ++t) se += __expf(lgt[t] - mx);
        mxse[0] = mx; mxse[1] = se;
    }
    __syncthreads();
    if (tid < 41) outp[b * 41 + tid] = __expf(lgt[tid] - mxse[0]) / mxse[1];
}

__global__ __launch_bounds__(256) void prior_head_kernel(
    const float* __restrict__ lat, const float* __restrict__ ps, const float* __restrict__ pb,
    const float* __restrict__ w1, const float* __restrict__ b1,
    const float* __restrict__ w2, const float* __restrict__ b2, float* __restrict__ outp)
{
    __shared__ float sh[256];
    __shared__ float ln[K_];
    __shared__ float red[256];
    __shared__ float hid[16];
    int b = blockIdx.x, tid = threadIdx.x;
    float v[4]; float s1 = 0.f, s2 = 0.f;
    #pragma unroll
    for (int i = 0; i < 4; ++i) {
        int k = tid + i * 256;
        v[i] = lat[(size_t)b * K_ + k];
        s1 += v[i]; s2 += v[i] * v[i];
    }
    float t1 = block_sum256(s1, sh), t2 = block_sum256(s2, sh);
    float mean = t1 * (1.f / K_), var = t2 * (1.f / K_) - mean * mean;
    float rs = rsqrtf(var + 1e-5f);
    #pragma unroll
    for (int i = 0; i < 4; ++i) {
        int k = tid + i * 256;
        ln[k] = (v[i] - mean) * rs * ps[k] + pb[k];
    }
    __syncthreads();
    int j = tid & 15, g = tid >> 4;
    float a = 0.f;
    for (int k = g * 64; k < g * 64 + 64; ++k) a += ln[k] * w1[k * 16 + j];
    red[tid] = a; __syncthreads();
    #pragma unroll
    for (int s = 8; s > 0; s >>= 1) {
        if (g < s) red[tid] += red[tid + s * 16];
        __syncthreads();
    }
    if (tid < 16) hid[tid] = gelu_fast(red[tid] + b1[tid]);
    __syncthreads();
    if (tid < 3) {
        float acc = b2[tid];
        #pragma unroll
        for (int jj = 0; jj < 16; ++jj) acc += hid[jj] * w2[jj * 3 + tid];
        float sg = sigmoid_fast(acc);
        const float smin = 0.1f;                     // 1/SMAX
        outp[b * 3 + tid] = smin + (10.0f - smin) * sg;
    }
}

// ---------------- KL sparsity ----------------
__global__ __launch_bounds__(256) void kl_kernel(const float* __restrict__ lat, float* __restrict__ kl_sum)
{
    __shared__ float sh[256];
    int k = blockIdx.x * 256 + threadIdx.x;          // < 1024
    float s = 0.f;
    for (int b = 0; b < B_; ++b) s += sigmoid_fast(lat[(size_t)b * K_ + k]);
    float rho = s * (1.f / B_);
    rho = fminf(fmaxf(rho, 1e-6f), 1.f - 1e-6f);
    const float t = 0.05f;
    float term = t * __logf(t / rho) + (1.f - t) * __logf((1.f - t) / (1.f - rho));
    float tot = block_sum256(term, sh);
    if (threadIdx.x == 0) atomicAdd(kl_sum, tot);
}

// ---------------- scalars ----------------
__global__ void init_scalars_kernel(float* scal) {
    if (threadIdx.x < 4) scal[threadIdx.x] = 0.f;
}
__global__ void finalize_kernel(const float* scal, float* o) {
    o[0] = scal[0] / ((float)C_ * B_ * T_ * D_);     // pc_recon
    o[1] = scal[1] / ((float)C_ * B_ * T_);          // pc_entropy
    o[2] = scal[2] * (1.f / K_);                     // kl
}

// ---------------- launch ----------------
extern "C" void kernel_launch(void* const* d_in, const int* in_sizes, int n_in,
                              void* d_out, int out_size, void* d_ws, size_t ws_size,
                              hipStream_t stream)
{
    const float* tokens = (const float*)d_in[0];
    const float* gln_s  = (const float*)d_in[1];
    const float* gln_b  = (const float*)d_in[2];
    const float* gw     = (const float*)d_in[3];
    const float* gb     = (const float*)d_in[4];
    const float* ew0    = (const float*)d_in[5];
    const float* eb0    = (const float*)d_in[6];
    const float* ew     = (const float*)d_in[7];
    const float* eb     = (const float*)d_in[8];
    const float* dw     = (const float*)d_in[9];
    const float* db     = (const float*)d_in[10];
    const float* oln_s  = (const float*)d_in[11];
    const float* oln_b  = (const float*)d_in[12];
    const float* ow1    = (const float*)d_in[13];
    const float* ob1    = (const float*)d_in[14];
    const float* ow2    = (const float*)d_in[15];
    const float* ob2    = (const float*)d_in[16];
    const float* pln_s  = (const float*)d_in[17];
    const float* pln_b  = (const float*)d_in[18];
    const float* pw1    = (const float*)d_in[19];
    const float* pb1    = (const float*)d_in[20];
    const float* pw2    = (const float*)d_in[21];
    const float* pb2    = (const float*)d_in[22];

    float* out = (float*)d_out;

    // ---- workspace layout (16B-aligned blocks) ----
    char* wsb = (char*)d_ws;
    float* err       = (float*)wsb;                        wsb += (size_t)MROWS * D_ * 4;   // [4096][2048] f32
    float* residuals = (float*)wsb;                        wsb += (size_t)C_ * B_ * D_ * 4; // [16][256][2048]
    float* z         = (float*)wsb;                        wsb += (size_t)C_ * B_ * K_ * 4; // [16][256][1024]
    float* latent    = (float*)wsb;                        wsb += (size_t)B_ * K_ * 4;      // [256][1024]
    float* scal      = (float*)wsb;                        wsb += 16;                       // sse, ent, kl
    unsigned short* tokb = (unsigned short*)wsb;           wsb += (size_t)MROWS * D_ * 2;   // tokens bf16
    unsigned short* h0b  = (unsigned short*)wsb;           wsb += (size_t)MROWS * K_ * 2;
    unsigned short* h1b  = (unsigned short*)wsb;           wsb += (size_t)MROWS * K_ * 2;
    unsigned short* wt0  = (unsigned short*)wsb;           wsb += (size_t)C_ * K_ * D_ * 2; // [c][K][D] bf16
    unsigned short* wth  = (unsigned short*)wsb;           wsb += (size_t)3 * C_ * K_ * K_ * 2; // [i][c][K][K]
    unsigned short* wtd  = (unsigned short*)wsb;           wsb += (size_t)C_ * D_ * K_ * 2; // [c][D][K]

    dim3 blk(256);

    init_scalars_kernel<<<1, 32, 0, stream>>>(scal);
    gate_kernel<<<B_, blk, 0, stream>>>(tokens, gln_s, gln_b, gw, gb, out + OUT_GATE);

    // pre-convert tokens; pre-transpose+convert all weights to bf16 [N][Kd]
    cvt_bf16_kernel<<<(MROWS * D_) / 1024, blk, 0, stream>>>(tokens, tokb);
    transpose_bf16_kernel<<<dim3(D_ / 32, K_ / 32, C_),     blk, 0, stream>>>(ew0, wt0, D_, K_);
    transpose_bf16_kernel<<<dim3(K_ / 32, K_ / 32, 3 * C_), blk, 0, stream>>>(ew,  wth, K_, K_);
    transpose_bf16_kernel<<<dim3(K_ / 32, D_ / 32, C_),     blk, 0, stream>>>(dw,  wtd, K_, D_);

    for (int c = 0; c < C_; ++c) {
        // encoder: tokens[4096x2048] @ ew0[c] -> gelu -> h0b (bf16)
        gemm_bf16_ep<0><<<dim3(K_ / BN, MROWS / BM), blk, 0, stream>>>(
            tokb, wt0 + (size_t)c * K_ * D_, eb0 + (size_t)c * K_, nullptr,
            h0b, nullptr, MROWS, K_, D_);
        // 3 hidden layers, ping-pong h0b<->h1b (final in h1b)
        for (int i = 0; i < 3; ++i) {
            const unsigned short* wi = wth + ((size_t)i * C_ + c) * K_ * K_;
            const float*          bi = eb + ((size_t)i * C_ + c) * K_;
            const unsigned short* src = (i % 2 == 0) ? h0b : h1b;
            unsigned short*       dst = (i % 2 == 0) ? h1b : h0b;
            gemm_bf16_ep<0><<<dim3(K_ / BN, MROWS / BM), blk, 0, stream>>>(
                src, wi, bi, nullptr, dst, nullptr, MROWS, K_, K_);
        }
        // decoder: h3[4096x1024] @ dw[c] + db - tokens -> err (f32)
        gemm_bf16_ep<1><<<dim3(D_ / BN, MROWS / BM), blk, 0, stream>>>(
            h1b, wtd + (size_t)c * D_ * K_, db + (size_t)c * D_, tokens,
            nullptr, err, MROWS, D_, K_);
        // reductions for this column
        reduce_err_kernel<<<(B_ * D_) / 256, blk, 0, stream>>>(err, residuals + (size_t)c * B_ * D_, scal + 0);
        zred_kernel<<<(B_ * K_) / 256, blk, 0, stream>>>(h1b, z + (size_t)c * B_ * K_);
        entropy_kernel<<<MROWS, blk, 0, stream>>>(h1b, scal + 1);
    }

    combine_residual_kernel<<<(B_ * D_) / 256, blk, 0, stream>>>(out + OUT_GATE, residuals, out + OUT_RES);
    combine_latent_kernel<<<(B_ * K_) / 256, blk, 0, stream>>>(out + OUT_GATE, z, latent);
    op_head_kernel<<<B_, blk, 0, stream>>>(latent, oln_s, oln_b, ow1, ob1, ow2, ob2, out + OUT_OP);
    prior_head_kernel<<<B_, blk, 0, stream>>>(latent, pln_s, pln_b, pw1, pb1, pw2, pb2, out + OUT_PRIOR);
    kl_kernel<<<K_ / 256, blk, 0, stream>>>(latent, scal + 2);
    finalize_kernel<<<1, 1, 0, stream>>>(scal, out + OUT_SCAL);
}
